// QuantizedLMHeadINT8_68461778698498
// MI455X (gfx1250) — matrix operand order repaired and checked by reading, not compiled, not run on hardware
//
#include <hip/hip_runtime.h>

// Quantized LM head: out[T,V] = (hidden[T,D] @ f32(W_int8[D,V])) * scale[V] + bias[V]
// HBM-bound: ~412MB weight (read ~once; L2-shared across m-tiles) + ~412MB out + 17MB
// hidden @ 23.3 TB/s => ~36us floor. WMMA just keeps compute free:
//  - v_wmma_f32_16x16x32_f16; hidden split hi/lo f16 (Markidis) into one fp32 C
//    (int8 weights exact in f16) => ~fp32 accuracy at 2x WMMA, ~10x under the roof.
//  - double-buffered LDS, one barrier/iter; next tile's global loads overlap WMMAs.
//  - weight edge: once-per-thread clamped column index (OOB cols never stored) =>
//    zero guard cost in the hot loop.
//  - weight staging: each thread owns a contiguous 16-row k-run so int8->f16 packs
//    into v_cvt_pk + aligned b32 LDS stores (8 instead of 16 b16).

typedef __attribute__((ext_vector_type(16))) _Float16 v16h;
typedef __attribute__((ext_vector_type(8)))  _Float16 v8h;
typedef __attribute__((ext_vector_type(4)))  _Float16 v4h;
typedef __attribute__((ext_vector_type(2)))  _Float16 v2h;
typedef __attribute__((ext_vector_type(8)))  float    v8f;

#define BM 128
#define BN 128
#define BK 32
#define BKP 40   // padded row stride in halves: 80B rows, 16B-aligned, conflict-free

__device__ __forceinline__ void stage_tile(_Float16* __restrict__ ah,
                                           _Float16* __restrict__ al,
                                           _Float16* __restrict__ bs,
                                           const float4* hbuf, const int* wreg,
                                           int hr0, int hc, int nc, int kb)
{
    #pragma unroll
    for (int q = 0; q < 4; ++q) {
        const float f[4] = { hbuf[q].x, hbuf[q].y, hbuf[q].z, hbuf[q].w };
        v4h hv, lv;
        #pragma unroll
        for (int e = 0; e < 4; ++e) {
            _Float16 h = (_Float16)f[e];
            hv[e] = h;
            lv[e] = (_Float16)(f[e] - (float)h);   // fp32 residual -> 2nd WMMA pass
        }
        *(v4h*)&ah[(hr0 + 32*q) * BKP + hc] = hv;  // 8B ds_store
        *(v4h*)&al[(hr0 + 32*q) * BKP + hc] = lv;
    }
    const int bbase = nc * BKP + 16 * kb;          // contiguous k-run of this thread
    #pragma unroll
    for (int qq = 0; qq < 16; qq += 2) {           // pack pairs -> aligned b32 stores
        v2h pk;
        pk[0] = (_Float16)wreg[qq];
        pk[1] = (_Float16)wreg[qq + 1];
        *(v2h*)&bs[bbase + qq] = pk;
    }
}

__device__ __forceinline__ void compute_tile(v8f acc[4][2],
                                             const _Float16* __restrict__ ah,
                                             const _Float16* __restrict__ al,
                                             const _Float16* __restrict__ bs,
                                             int wm, int wn, int l16, int kh)
{
    v16h bfr[2];
    #pragma unroll
    for (int j = 0; j < 2; ++j) { // B frag: halves [n][8*kh..+7] and [n][16+8*kh..+7]
        const _Float16* rb = &bs[(wn*32 + j*16 + l16) * BKP + 8*kh];
        bfr[j] = __builtin_shufflevector(*(const v8h*)rb, *(const v8h*)(rb + 16), 0,1,2,3,4,5,6,7,8,9,10,11,12,13,14,15);
    }

    #pragma unroll
    for (int i = 0; i < 4; ++i) {
        const _Float16* ra = &ah[(wm*64 + i*16 + l16) * BKP + 8*kh];
        v8h a0 = *(const v8h*)ra;                  // K = 8*kh..+7     -> VGPR 0..3
        v8h a1 = *(const v8h*)(ra + 16);           // K = 16+8*kh..+7  -> VGPR 4..7
        v16h ahf = __builtin_shufflevector(a0, a1, 0,1,2,3,4,5,6,7,8,9,10,11,12,13,14,15);
        const _Float16* rl = &al[(wm*64 + i*16 + l16) * BKP + 8*kh];
        v8h c0 = *(const v8h*)rl;
        v8h c1 = *(const v8h*)(rl + 16);
        v16h alf = __builtin_shufflevector(c0, c1, 0,1,2,3,4,5,6,7,8,9,10,11,12,13,14,15);
        #pragma unroll
        for (int j = 0; j < 2; ++j) {
            acc[i][j] = __builtin_amdgcn_wmma_f32_16x16x32_f16(
                false, ahf, false, bfr[j], (short)0, acc[i][j], false, false);
            acc[i][j] = __builtin_amdgcn_wmma_f32_16x16x32_f16(
                false, alf, false, bfr[j], (short)0, acc[i][j], false, false);
        }
    }
}

__global__ __launch_bounds__(256)
void QuantizedLMHeadINT8_wmma_kernel(const float* __restrict__ hidden,   // [T,D] f32
                                     const int*   __restrict__ weight,   // [D,V] int8-in-int32
                                     const float* __restrict__ scale,    // [V]
                                     const float* __restrict__ bias,     // [V]
                                     float*       __restrict__ out,      // [T,V]
                                     int T, int D, int V)
{
    __shared__ _Float16 As_hi[2][BM * BKP];   // 2 x 10240 B
    __shared__ _Float16 As_lo[2][BM * BKP];
    __shared__ _Float16 Bs   [2][BN * BKP];   // [n][k] transposed

    const int tid  = threadIdx.x;
    const int lane = tid & 31;
    const int wid  = tid >> 5;      // 8 waves (wave32)
    const int wm   = wid >> 2;      // 0..1 : 64-row band
    const int wn   = wid & 3;       // 0..3 : 32-col band
    const int l16  = lane & 15;
    const int kh   = lane >> 4;

    const int bm0 = blockIdx.x * BM;
    const int bn0 = blockIdx.y * BN;

    // ---- k-invariant staging geometry (hoisted: zero guard cost in the loop) ----
    const int hc  = (tid & 7) * 4;  // hidden: thread owns one float4 column
    const int hr0 = tid >> 3;       //         rows hr0 + 32q
    const float* hptr[4];
    #pragma unroll
    for (int q = 0; q < 4; ++q) {
        int gm = bm0 + hr0 + 32*q; if (gm >= T) gm = T - 1;
        hptr[q] = hidden + (size_t)gm * D + hc;
    }
    const int nc = tid & 127;       // weight: thread owns one column (coalesced b32)
    const int kb = tid >> 7;        //         contiguous k-rows 16*kb + qq
    int gn_w = bn0 + nc; if (gn_w >= V) gn_w = V - 1;  // clamp addr only; col never stored
    const int* wptr = weight + (size_t)(16 * kb) * V + gn_w;
    const size_t wstep = (size_t)BK * V;    // next k-tile
    const size_t wrow  = (size_t)V;         // between this thread's k-rows

    v8f acc[4][2] = {};
    const int NK = D / BK;          // D assumed multiple of BK (2048/32)

    // ---------------- prologue: fetch + stage tile 0 ----------------
    float4 hbuf[4];
    int    wreg[16];
    #pragma unroll
    for (int q = 0; q < 4; ++q) hbuf[q] = *(const float4*)hptr[q];
    #pragma unroll
    for (int qq = 0; qq < 16; ++qq) wreg[qq] = wptr[(size_t)qq * wrow];
    stage_tile(As_hi[0], As_lo[0], Bs[0], hbuf, wreg, hr0, hc, nc, kb);
    __syncthreads();

    // ---------------- main pipeline: 1 barrier / iteration ----------------
    int p = 0;
    for (int it = 0; it < NK; ++it) {
        if (it + 1 < NK) {          // issue next tile's global loads before WMMAs
            #pragma unroll
            for (int q = 0; q < 4; ++q) { hptr[q] += BK; hbuf[q] = *(const float4*)hptr[q]; }
            wptr += wstep;
            #pragma unroll
            for (int qq = 0; qq < 16; ++qq) wreg[qq] = wptr[(size_t)qq * wrow];
            __builtin_prefetch((const char*)(wptr + wstep), 0, 1);  // k+2 weight slab
        }
        compute_tile(acc, As_hi[p], As_lo[p], Bs[p], wm, wn, l16, kh);
        if (it + 1 < NK)
            stage_tile(As_hi[p ^ 1], As_lo[p ^ 1], Bs[p ^ 1], hbuf, wreg, hr0, hc, nc, kb);
        __syncthreads();
        p ^= 1;
    }

    // ---------------- epilogue: per-column scale + bias ----------------
    #pragma unroll
    for (int j = 0; j < 2; ++j) {
        int gn = bn0 + wn*32 + j*16 + l16;
        bool okn = gn < V;
        float s  = okn ? scale[gn] : 0.f;
        float bb = okn ? bias[gn]  : 0.f;
        #pragma unroll
        for (int i = 0; i < 4; ++i) {
            #pragma unroll
            for (int r = 0; r < 8; ++r) {   // C frag: row = r + 8*kh, col = l16
                int gm = bm0 + wm*64 + i*16 + 8*kh + r;
                if (okn && gm < T)
                    out[(size_t)gm * V + gn] = acc[i][j][r] * s + bb;
            }
        }
    }
}

extern "C" void kernel_launch(void* const* d_in, const int* in_sizes, int n_in,
                              void* d_out, int out_size, void* d_ws, size_t ws_size,
                              hipStream_t stream) {
    const float* hidden = (const float*)d_in[0];   // [B*T, D] f32
    const int*   weight = (const int*)  d_in[1];   // [D, V]   int32 carrier
    const float* scale  = (const float*)d_in[2];   // [V]
    const float* bias   = (const float*)d_in[3];   // [V]
    float*       out    = (float*)d_out;           // [B*T, V] f32

    const int V = in_sizes[2];
    const int D = in_sizes[1] / V;
    const int T = in_sizes[0] / D;                 // B*T rows

    // m-tiles on x (fast) so concurrent WGs share one weight n-panel via L2
    dim3 grid((T + BM - 1) / BM, (V + BN - 1) / BN);
    QuantizedLMHeadINT8_wmma_kernel<<<grid, 256, 0, stream>>>(
        hidden, weight, scale, bias, out, T, D, V);
}
